// BasisDecentralizedModel_33277406609823
// MI455X (gfx1250) — matrix operand order repaired and checked by reading, not compiled, not run on hardware
//
#include <hip/hip_runtime.h>
#include <math.h>

typedef __attribute__((ext_vector_type(16))) _Float16 v16h;
typedef __attribute__((ext_vector_type(8)))  float    v8f;

// rot90 applied r times (numpy convention, counterclockwise) on NxN:
// returns source indices (si,sj) such that rot90(A, r)[i][j] == A[si][sj]
__device__ __forceinline__ void rotsrc(int r, int i, int j, int N, int& si, int& sj) {
  switch (r & 3) {
    case 0:  si = i;       sj = j;       break;
    case 1:  si = j;       sj = N-1-i;   break;
    case 2:  si = N-1-i;   sj = N-1-j;   break;
    default: si = N-1-j;   sj = i;       break;
  }
}

// ---------------------------------------------------------------------------
// Prep kernels: bake rotated weights into WMMA B-fragment-major layout.
// B tile (32K x 16N) fragment order: buf[tile*512 + lane*16 + e], where
// lane = kh*16 + n (kh = lane>>4, n = lane&15) and element e maps to
// k = kh*16 + e. Each lane's 16 f16 are contiguous (32B -> global_load_b128).
// ---------------------------------------------------------------------------

// feat zero + W4g[o][h][a] = W4[o][(h-a)%2]
__global__ void prep_misc(const float* __restrict__ W4,
                          unsigned* __restrict__ feat,
                          float* __restrict__ W4g) {
  int idx = blockIdx.x * 256 + threadIdx.x;
  if (idx < 8192) feat[idx] = 0u;
  if (idx < 2048) {
    int o = idx >> 3, h = (idx >> 1) & 3, a = idx & 1;
    W4g[idx] = W4[o * 2 + ((h - a) & 1)];
  }
}

// K1[co=c1*4+r][ci][ky][kx] = rot90(W1[c1][ci], r)[ky][kx], K order k=ci*64+ky*8+kx
__global__ void prep_b1(const float* __restrict__ W1, _Float16* __restrict__ B1) {
  int idx = blockIdx.x * 256 + threadIdx.x;
  if (idx >= 6144) return;
  int e = idx & 15, lane = (idx >> 4) & 31, t = idx >> 9;   // t in 0..11
  int nt = t / 6, s = t - nt * 6;
  int n = lane & 15, kh = lane >> 4;
  int k = s * 32 + kh * 16 + e;                              // 0..191
  int co = nt * 16 + n, c1 = co >> 2, r = co & 3;
  int ci = k >> 6, rem = k & 63, ky = rem >> 3, kx = rem & 7;
  int sy, sx; rotsrc(r, ky, kx, 8, sy, sx);
  B1[idx] = (_Float16)W1[((c1 * 3 + ci) * 8 + sy) * 8 + sx];
}

// K2[co2=c16*4+r][ci2=ci8*4+gi][ky][kx] = rot90(W2[c16][ci8][(gi-r)%4], r)[ky][kx]
// K order k = (ky*5+kx)*32 + ci2  (channel fastest)
__global__ void prep_b2(const float* __restrict__ W2, _Float16* __restrict__ B2) {
  int idx = blockIdx.x * 256 + threadIdx.x;
  if (idx >= 51200) return;
  int e = idx & 15, lane = (idx >> 4) & 31, t = idx >> 9;   // t in 0..99
  int nt = t / 25, s = t - nt * 25;
  int n = lane & 15, kh = lane >> 4;
  int k = s * 32 + kh * 16 + e;                              // 0..799
  int ci2 = k & 31, pix = k >> 5, ky = pix / 5, kx = pix - ky * 5;
  int co2 = nt * 16 + n, c16 = co2 >> 2, r = co2 & 3;
  int ci8 = ci2 >> 2, gi = ci2 & 3, gs = (gi - r) & 3;
  int sy, sx; rotsrc(r, ky, kx, 5, sy, sx);
  B2[idx] = (_Float16)W2[(((c16 * 8 + ci8) * 4 + gs) * 5 + sy) * 5 + sx];
}

// Wgh[o][c][h][g] = Wg[o][c][(h-g)%4], K order k = c*4+g, per-h B buffers
__global__ void prep_bh(const float* __restrict__ Wg, _Float16* __restrict__ BH) {
  int idx = blockIdx.x * 256 + threadIdx.x;
  if (idx >= 131072) return;
  int e = idx & 15, lane = (idx >> 4) & 31;
  int h = idx >> 15, t = (idx >> 9) & 63;                    // t in 0..63
  int nt = t >> 2, s = t & 3;
  int n = lane & 15, kh = lane >> 4;
  int k = s * 32 + kh * 16 + e;                              // 0..127
  int c = k >> 2, g = k & 3;
  int o = nt * 16 + n;
  BH[idx] = (_Float16)Wg[(o * 32 + c) * 4 + ((h - g) & 3)];
}

// ---------------------------------------------------------------------------
// Fused conv1 -> relu -> conv2 -> relu -> tile max  (per 16x16 conv2 tile).
// A-fragment element e of lane (m=lane&15, kh=lane>>4) maps to
// k = kh*16 + e  within the 32-wide K step; K orderings are
// chosen so each fragment half is 8 contiguous f16 (merges to ds_load_b128).
// Each A gather feeds ALL N-tiles (2 for conv1, 4 for conv2) to cut LDS
// traffic per WMMA 2-4x.
// ---------------------------------------------------------------------------
__global__ __launch_bounds__(256) void conv_fused(
    const float* __restrict__ states, const float* __restrict__ b1,
    const float* __restrict__ b2, const _Float16* __restrict__ gB1,
    const _Float16* __restrict__ gB2, unsigned* __restrict__ gFeat) {
  __shared__ _Float16 sIn[3 * 27 * 27];     // input patch  [ci][y][x]
  __shared__ _Float16 sP1[20 * 20 * 32];    // conv1 patch  [y][x][co]
  __shared__ unsigned sMax[64];

  const int tid = threadIdx.x, lane = tid & 31;
  const int wave = __builtin_amdgcn_readfirstlane(tid >> 5);  // SGPR wave id
  const int n = blockIdx.x;                       // image 0..127
  const int ty = blockIdx.y / 6, tx = blockIdx.y - 6 * (blockIdx.y / 6);
  const int oy0 = ty * 16, ox0 = tx * 16;

  if (tid < 64) sMax[tid] = 0u;
  // warm L2/L0 with the (shared, cache-resident) swizzled weights
  for (int off = tid * 128; off < 51200 * 2; off += 256 * 128)
    __builtin_prefetch((const char*)gB2 + off, 0, 3);
  __builtin_prefetch((const char*)gB1 + (tid & 127) * 96, 0, 3);

  // stage 27x27x3 input patch (zero-padded beyond 100x100)
  for (int idx = tid; idx < 3 * 27 * 27; idx += 256) {
    int ci = idx / 729, rem = idx - ci * 729;
    int py = rem / 27, px = rem - py * 27;
    int gy = oy0 + py, gx = ox0 + px;
    float v = (gy < 100 && gx < 100)
                  ? states[((n * 3 + ci) * 100 + gy) * 100 + gx] : 0.f;
    sIn[idx] = (_Float16)v;
  }
  __syncthreads();

  const int m = lane & 15, kh = lane >> 4;

  // ---- conv1: M=400 (25 tiles), N=32 (2 tiles), K=192 (6 steps) ----
  for (int mt = wave; mt < 25; mt += 8) {          // scalar loop (SGPR wave)
    int p = mt * 16 + m, py = p / 20, px = p - py * 20;
    v8f acc0 = {}, acc1 = {};
    for (int s = 0; s < 6; ++s) {
      v16h af;
      #pragma unroll
      for (int e = 0; e < 16; ++e) {
        int k = s * 32 + kh * 16 + e;
        int ci = k >> 6, rem = k & 63, ky = rem >> 3, kx = rem & 7;
        af[e] = sIn[ci * 729 + (py + ky) * 27 + (px + kx)];
      }
      v16h bf0 = *(const v16h*)(gB1 + (s * 32 + lane) * 16);
      v16h bf1 = *(const v16h*)(gB1 + ((6 + s) * 32 + lane) * 16);
      acc0 = __builtin_amdgcn_wmma_f32_16x16x32_f16(false, af, false, bf0,
                                                    (short)0, acc0, false, false);
      acc1 = __builtin_amdgcn_wmma_f32_16x16x32_f16(false, af, false, bf1,
                                                    (short)0, acc1, false, false);
    }
    float bias0 = b1[m >> 2], bias1 = b1[(16 + m) >> 2];
    #pragma unroll
    for (int r = 0; r < 8; ++r) {
      int pp = mt * 16 + r + kh * 8;
      int ppy = pp / 20, ppx = pp - ppy * 20;
      int base = (ppy * 20 + ppx) * 32;
      sP1[base + m]      = (_Float16)fmaxf(acc0[r] + bias0, 0.f);
      sP1[base + 16 + m] = (_Float16)fmaxf(acc1[r] + bias1, 0.f);
    }
  }
  __syncthreads();

  // ---- conv2: M=256 (16 tiles), N=64 (4 tiles), K=800 (25 steps) + max ----
  {
    float mv[4] = {0.f, 0.f, 0.f, 0.f};
    for (int half = 0; half < 2; ++half) {
      int mt = wave + half * 8;                    // scalar
      int p = mt * 16 + m, py = p >> 4, px = p & 15;
      v8f acc[4] = {};
      for (int s = 0; s < 25; ++s) {
        v16h af;
        #pragma unroll
        for (int e = 0; e < 16; ++e) {
          int k = s * 32 + kh * 16 + e;
          int ci2 = k & 31, pix = k >> 5, ky = pix / 5, kx = pix - ky * 5;
          af[e] = sP1[((py + ky) * 20 + (px + kx)) * 32 + ci2];
        }
        #pragma unroll
        for (int nt = 0; nt < 4; ++nt) {
          v16h bf = *(const v16h*)(gB2 + ((nt * 25 + s) * 32 + lane) * 16);
          acc[nt] = __builtin_amdgcn_wmma_f32_16x16x32_f16(
              false, af, false, bf, (short)0, acc[nt], false, false);
        }
      }
      #pragma unroll
      for (int nt = 0; nt < 4; ++nt) {
        float bias = b2[(nt * 16 + m) >> 2];
        #pragma unroll
        for (int r = 0; r < 8; ++r) {
          int pp = mt * 16 + r + kh * 8;
          int ppy = pp >> 4, ppx = pp & 15;
          if (oy0 + ppy < 89 && ox0 + ppx < 89)
            mv[nt] = fmaxf(mv[nt], fmaxf(acc[nt][r] + bias, 0.f));
        }
      }
    }
    #pragma unroll
    for (int nt = 0; nt < 4; ++nt) {
      float v = fmaxf(mv[nt], __shfl_xor(mv[nt], 16, 32));
      if (lane < 16) atomicMax(&sMax[nt * 16 + m], __float_as_uint(v));
    }
  }
  __syncthreads();
  if (tid < 64) atomicMax(&gFeat[n * 64 + tid], sMax[tid]);  // relu => >=0
}

// ---------------------------------------------------------------------------
// Head: norms/agg -> z (LDS f16), z x Wgh per h via WMMA (M=128,N=256,K=128),
// fused relu + policy/value contraction + bias epilogue. Each wave holds the
// 4 A fragments of its (h,mt) pair in VGPRs and streams 16 N-tiles past them.
// ---------------------------------------------------------------------------
__global__ __launch_bounds__(256) void head_kernel(
    const float* __restrict__ diffs, const float* __restrict__ bg,
    const float* __restrict__ b4, const float* __restrict__ W5,
    const float* __restrict__ b5, const float* __restrict__ gFeat,
    const _Float16* __restrict__ gBH, const float* __restrict__ gW4g,
    float* __restrict__ out) {
  __shared__ _Float16 sZ[128 * 128];   // [row=b*4+i][k=c*4+g]
  __shared__ float sN[512];            // norms [b][i][j]
  __shared__ float sPol[256];
  __shared__ float sVal[128];

  const int tid = threadIdx.x, lane = tid & 31;
  const int wave = __builtin_amdgcn_readfirstlane(tid >> 5);
  if (tid < 128) { sVal[tid] = 0.f; sPol[tid] = 0.f; sPol[128 + tid] = 0.f; }
  for (int off = tid * 128; off < 131072 * 2; off += 256 * 128)
    __builtin_prefetch((const char*)gBH + off, 0, 3);
  for (int idx = tid; idx < 512; idx += 256) {
    int b = idx >> 4, i = (idx >> 2) & 3, j = idx & 3;
    float dx = diffs[(b * 4 + i) * 2 + 0] - diffs[(b * 4 + j) * 2 + 0];
    float dy = diffs[(b * 4 + i) * 2 + 1] - diffs[(b * 4 + j) * 2 + 1];
    sN[idx] = sqrtf(dx * dx + dy * dy + 1e-12f);
  }
  __syncthreads();
  for (int idx = tid; idx < 128 * 128; idx += 256) {
    int row = idx >> 7, kk = idx & 127;
    int c = kk >> 2, g = kk & 3;
    int b = row >> 2, i = row & 3;
    float v;
    if (c < 16) {
      v = gFeat[row * 64 + c * 4 + g];
    } else {
      int c2 = c - 16; v = 0.f;
      #pragma unroll
      for (int j = 0; j < 4; ++j)
        v += sN[b * 16 + i * 4 + j] * gFeat[(b * 4 + j) * 64 + c2 * 4 + g];
    }
    sZ[idx] = (_Float16)v;
  }
  __syncthreads();

  const int m = lane & 15, kh = lane >> 4;
  for (int q = 0; q < 4; ++q) {
    int pi = wave * 4 + q;                       // (h, mt) pair, scalar
    int h = pi >> 3, mt = pi & 7;
    v16h afs[4];                                 // all K fragments of this M-tile
    #pragma unroll
    for (int s = 0; s < 4; ++s) {
      #pragma unroll
      for (int e = 0; e < 16; ++e) {
        int k = s * 32 + kh * 16 + e;
        afs[s][e] = sZ[(mt * 16 + m) * 128 + k];
      }
    }
    for (int ng = 0; ng < 4; ++ng) {             // 4 groups of 4 N-tiles
      v8f acc[4] = {};
      #pragma unroll
      for (int s = 0; s < 4; ++s) {
        #pragma unroll
        for (int j = 0; j < 4; ++j) {
          int nt = ng * 4 + j;
          v16h bf = *(const v16h*)(gBH + (((h * 16 + nt) * 4 + s) * 32 + lane) * 16);
          acc[j] = __builtin_amdgcn_wmma_f32_16x16x32_f16(
              false, afs[s], false, bf, (short)0, acc[j], false, false);
        }
      }
      #pragma unroll
      for (int j = 0; j < 4; ++j) {
        int o = (ng * 4 + j) * 16 + m;
        float bias = bg[o], w5v = W5[o];
        float w40 = gW4g[(o * 4 + h) * 2 + 0], w41 = gW4g[(o * 4 + h) * 2 + 1];
        #pragma unroll
        for (int r = 0; r < 8; ++r) {
          int row = mt * 16 + r + kh * 8;
          float hv = fmaxf(acc[j][r] + bias, 0.f);
          atomicAdd(&sVal[row], hv * w5v);
          atomicAdd(&sPol[row * 2 + 0], hv * w40);
          atomicAdd(&sPol[row * 2 + 1], hv * w41);
        }
      }
    }
  }
  __syncthreads();
  if (tid < 128) {
    out[tid * 2 + 0] = sPol[tid * 2 + 0] + b4[0];   // policies (32,4,2) flat
    out[tid * 2 + 1] = sPol[tid * 2 + 1] + b4[1];
    out[256 + tid]   = sVal[tid] + b5[0];           // values (32,4) flat
  }
}

extern "C" void kernel_launch(void* const* d_in, const int* in_sizes, int n_in,
                              void* d_out, int out_size, void* d_ws, size_t ws_size,
                              hipStream_t stream) {
  const float* diffs  = (const float*)d_in[0];
  const float* states = (const float*)d_in[1];
  const float* W1     = (const float*)d_in[2];
  const float* b1     = (const float*)d_in[3];
  const float* W2     = (const float*)d_in[4];
  const float* b2     = (const float*)d_in[5];
  const float* Wg     = (const float*)d_in[6];
  const float* bg     = (const float*)d_in[7];
  const float* W4     = (const float*)d_in[8];
  const float* b4     = (const float*)d_in[9];
  const float* W5     = (const float*)d_in[10];
  const float* b5     = (const float*)d_in[11];
  (void)in_sizes; (void)n_in; (void)out_size; (void)ws_size;

  char* ws = (char*)d_ws;
  unsigned*  feat = (unsigned*)(ws + 0);        // 8192 u32  (32 KB)
  _Float16*  B1   = (_Float16*)(ws + 32768);    // 6144 f16
  _Float16*  B2   = (_Float16*)(ws + 45056);    // 51200 f16
  _Float16*  BH   = (_Float16*)(ws + 147456);   // 131072 f16
  float*     W4g  = (float*)   (ws + 409600);   // 2048 f32   -> ~408 KB total

  prep_misc<<<32,  256, 0, stream>>>(W4, feat, W4g);
  prep_b1  <<<24,  256, 0, stream>>>(W1, B1);
  prep_b2  <<<200, 256, 0, stream>>>(W2, B2);
  prep_bh  <<<512, 256, 0, stream>>>(Wg, BH);
  conv_fused<<<dim3(128, 36), 256, 0, stream>>>(states, b1, b2, B1, B2, feat);
  head_kernel<<<1, 256, 0, stream>>>(diffs, bg, b4, W5, b5,
                                     (const float*)feat, BH, W4g, (float*)d_out);
}
